// CRF_11132555231302
// MI455X (gfx1250) — matrix operand order repaired and checked by reading, not compiled, not run on hardware
//
#include <hip/hip_runtime.h>
#include <hip/hip_bf16.h>

// CRF log-likelihood for MI455X (gfx1250, wave32).
//
// Forward step  new_alpha[j] = logsumexp_i(alpha[i]+trans[i,j]) + emit[j]
// rewritten with a per-row offset as a plus-times GEMM:
//   new_alpha[j] = g + log( sum_i exp(alpha[i]-g) * E[i,j] ) + emit[j],
// E = exp(trans) constant -> per step a (256x128)x(128x128) f16 GEMM on
// v_wmma_f32_16x16x32_f16 (two independent accumulator chains of depth 2).
//
// Latency-bound by the 1024-step scan; per-step critical path:
//  - ONE barrier per step: offset lagged one step (g(t) = alpha(t-1)[row][0]+C),
//    sh_exp/sh_off double-buffered, loop unrolled x2 with static ping-pong,
//  - split s_barrier_signal/-wait, next-step global prefetch in the window,
//  - all-finite math (eps in log, clamped exp2) -> mask applied as fma, no
//    exec branches; raw v_log_f32 / v_exp_f32,
//  - sched_group_barrier hints: issue all 8 A-fragment ds_load_b128 before the
//    WMMA chain (one pipelined LDS latency instead of four serial ones).

typedef _Float16 v16h __attribute__((ext_vector_type(16)));
typedef _Float16 v8h  __attribute__((ext_vector_type(8)));
typedef float    v8f  __attribute__((ext_vector_type(8)));
typedef float    v4f  __attribute__((ext_vector_type(4)));

#define B_   256
#define S_   1024
#define T_   97
#define TP   128      // padded tag count
#define NW   7        // 7 N-tiles of 16 cover 112 >= 97
#define STRIDE 136    // sh_exp row stride in halves (272B: 16B aligned, bank-skewed)
#define EPS_  1.0e-35f
#define LOG2E 1.4426950408889634f
#define LN2   0.6931471805599453f
#define GLAG  4.5f    // expected per-step logZ growth (~log 97)

// split workgroup barrier: dscnt-flush + signal, then wait (prefetch in window)
#define BAR_SIGNAL() asm volatile("s_wait_dscnt 0x0\n\ts_barrier_signal -1" ::: "memory")
#define BAR_WAIT()   asm volatile("s_barrier_wait -1" ::: "memory")

// scheduling pipeline hint: 8 DS reads (A frags), 4 WMMA, 2 DS reads (offsets)
#if __has_builtin(__builtin_amdgcn_sched_group_barrier)
#define SCHED_PIPELINE()                                   \
  do {                                                     \
    __builtin_amdgcn_sched_group_barrier(0x100, 8, 0);     \
    __builtin_amdgcn_sched_group_barrier(0x008, 4, 0);     \
    __builtin_amdgcn_sched_group_barrier(0x100, 2, 0);     \
  } while (0)
#define SCHED_PIPELINE_PEEL()                              \
  do {                                                     \
    __builtin_amdgcn_sched_group_barrier(0x100, 8, 0);     \
    __builtin_amdgcn_sched_group_barrier(0x008, 4, 0);     \
  } while (0)
#else
#define SCHED_PIPELINE()      do {} while (0)
#define SCHED_PIPELINE_PEEL() do {} while (0)
#endif

// ---------------------------------------------------------------------------
// Kernel 1: ET[j][i] = exp(transitions[i][j]) as f16, 128x128, zero padded.
// ---------------------------------------------------------------------------
__global__ __launch_bounds__(256) void crf_prep_et(const float* __restrict__ trans,
                                                   _Float16* __restrict__ et) {
    int idx = blockIdx.x * blockDim.x + threadIdx.x;
    if (idx >= TP * TP) return;
    int j = idx >> 7, i = idx & (TP - 1);
    float v = 0.0f;
    if (i < T_ && j < T_) v = __expf(trans[i * T_ + j]);
    et[idx] = (_Float16)v;
}

// One scan step. SRCE/DSTE: sh_exp buffers; SRCO/DSTO: sh_off buffers;
// EC/MC: current emissions/mask; EN/MN: next (prefetched); GP: offset used for
// V(t-1); GN: offset for V(t) (lagged column-0 value read from SRCO).
#define STEP_BODY(T, SRCE, DSTE, SRCO, DSTO, EC, MC, EN, MN, GP, GN)           \
  do {                                                                         \
    BAR_WAIT();                                                                \
    const _Float16* abase = &(SRCE)[n * STRIDE + sub * 8];                     \
    FragU a0, a1, a2, a3;                                                      \
    a0.h[0] = *(const v8h*)(abase);       a0.h[1] = *(const v8h*)(abase + 16); \
    a1.h[0] = *(const v8h*)(abase + 32);  a1.h[1] = *(const v8h*)(abase + 48); \
    a2.h[0] = *(const v8h*)(abase + 64);  a2.h[1] = *(const v8h*)(abase + 80); \
    a3.h[0] = *(const v8h*)(abase + 96);  a3.h[1] = *(const v8h*)(abase + 112);\
    v8f zz0 = {}, zz1 = {};                                                    \
    v8f acc0 = __builtin_amdgcn_wmma_f32_16x16x32_f16(                         \
        false, a0.v, false, bfrag[0].v, (short)0, zz0, false, false);          \
    v8f acc1 = __builtin_amdgcn_wmma_f32_16x16x32_f16(                         \
        false, a1.v, false, bfrag[1].v, (short)0, zz1, false, false);          \
    acc0 = __builtin_amdgcn_wmma_f32_16x16x32_f16(                             \
        false, a2.v, false, bfrag[2].v, (short)0, acc0, false, false);         \
    acc1 = __builtin_amdgcn_wmma_f32_16x16x32_f16(                             \
        false, a3.v, false, bfrag[3].v, (short)0, acc1, false, false);         \
    v4f po0 = *(const v4f*)&(SRCO)[8 * sub];                                   \
    v4f po1 = *(const v4f*)&(SRCO)[8 * sub + 4];                               \
    SCHED_PIPELINE();                                                          \
    _Pragma("unroll")                                                          \
    for (int v = 0; v < 8; ++v) {                                              \
      float lg    = __builtin_amdgcn_logf(acc0[v] + acc1[v] + EPS_);           \
      float fresh = fmaf(LN2, lg, (GP)[v] + (EC)[v]);                          \
      alpha[v]    = fmaf((MC)[v], fresh - alpha[v], alpha[v]);                 \
    }                                                                          \
    _Pragma("unroll")                                                          \
    for (int v = 0; v < 8; ++v)                                                \
      (GN)[v] = ((v < 4) ? po0[v & 3] : po1[v & 3]) + GLAG;                    \
    if (w == 0 && n == 0) {                                                    \
      v4f b0v = { alpha[0], alpha[1], alpha[2], alpha[3] };                    \
      v4f b1v = { alpha[4], alpha[5], alpha[6], alpha[7] };                    \
      *(v4f*)&(DSTO)[8 * sub]     = b0v;                                       \
      *(v4f*)&(DSTO)[8 * sub + 4] = b1v;                                       \
    }                                                                          \
    _Pragma("unroll")                                                          \
    for (int v = 0; v < 8; ++v) {                                              \
      float d = fminf(LOG2E * (alpha[v] - (GN)[v]), 15.9f);                    \
      (DSTE)[(v + 8 * sub) * STRIDE + j] = (_Float16)__builtin_amdgcn_exp2f(d);\
    }                                                                          \
    BAR_SIGNAL();                                                              \
    _Pragma("unroll")                                                          \
    for (int v = 0; v < 8; ++v) {                                              \
      int m = v + 8 * sub;                                                     \
      (EN)[v] = inputs[((size_t)(b0 + m) * S_ + (T) + 1) * T_ + je];           \
      (MN)[v] = (float)mask[(size_t)(b0 + m) * S_ + (T) + 1];                  \
    }                                                                          \
  } while (0)

// ---------------------------------------------------------------------------
// Kernel 2: forward scan (log-denominator). One block = 16 batch rows,
// 7 waves = 7 N-tiles of 16 tag-columns. Adds -logZ_b into out[0].
// ---------------------------------------------------------------------------
__global__ __launch_bounds__(NW * 32, 1) void crf_forward(
    const float* __restrict__ inputs,    // [B,S,T]
    const float* __restrict__ start_t,   // [T]
    const float* __restrict__ end_t,     // [T]
    const int*   __restrict__ mask,      // [B,S]
    const _Float16* __restrict__ et,     // [128,128] = exp(trans)^T, padded
    float* __restrict__ out) {
    __shared__ __align__(16) _Float16 sh_exp[2][16 * STRIDE]; // double buffered
    __shared__ __align__(16) float    sh_off[2][16];
    __shared__ float                  sh_red[16 * 8];

    const int tid  = threadIdx.x;
    const int w    = tid >> 5;
    const int lane = tid & 31;
    const int n    = lane & 15;
    const int sub  = lane >> 4;
    const int j    = w * 16 + n;
    const bool jok = (j < T_);
    const int je   = jok ? j : 0;
    const int b0   = blockIdx.x * 16;
    const float NEG_INF = -__builtin_inff();

    // one-time: zero never-written columns 112..135 in both buffers (no NaNs)
    for (int x = tid; x < 16 * STRIDE; x += blockDim.x)
        if ((x % STRIDE) >= 112) {
            sh_exp[0][x] = (_Float16)0.0f;
            sh_exp[1][x] = (_Float16)0.0f;
        }

    union FragU { v16h v; v8h h[2]; };
    FragU bfrag[4];
#pragma unroll
    for (int q = 0; q < 4; ++q) {
        const _Float16* p = et + (size_t)j * TP + q * 32 + sub * 8;
        bfrag[q].h[0] = *(const v8h*)(p);
        bfrag[q].h[1] = *(const v8h*)(p + 16);
    }

    float evl = end_t[je];
    const float endv = jok ? evl : NEG_INF;

    // ---- t = 0: alpha0 = start + emit0 ----
    float alpha[8];
    float eA[8], mA[8], eB[8], mB[8], gA[8], gB[8];
    {
        float sj = start_t[je];
#pragma unroll
        for (int v = 0; v < 8; ++v) {
            int m = v + 8 * sub;
            alpha[v] = sj + inputs[((size_t)(b0 + m) * S_) * T_ + je];
        }
    }
    if (w == 0 && n == 0) {
        v4f a0 = { alpha[0], alpha[1], alpha[2], alpha[3] };
        v4f a1 = { alpha[4], alpha[5], alpha[6], alpha[7] };
        *(v4f*)&sh_off[0][8 * sub]     = a0;
        *(v4f*)&sh_off[0][8 * sub + 4] = a1;
    }
    BAR_SIGNAL();
#pragma unroll
    for (int v = 0; v < 8; ++v) {       // prefetch t=1 inside the window
        int m = v + 8 * sub;
        eA[v] = inputs[((size_t)(b0 + m) * S_ + 1) * T_ + je];
        mA[v] = (float)mask[(size_t)(b0 + m) * S_ + 1];
    }
    BAR_WAIT();
    {
        v4f o0 = *(const v4f*)&sh_off[0][8 * sub];
        v4f o1 = *(const v4f*)&sh_off[0][8 * sub + 4];
#pragma unroll
        for (int v = 0; v < 8; ++v) {
            gA[v] = (v < 4) ? o0[v & 3] : o1[v & 3];   // exact at t=0
            float d = fminf(LOG2E * (alpha[v] - gA[v]), 15.9f);
            sh_exp[0][(v + 8 * sub) * STRIDE + j] =
                (_Float16)__builtin_amdgcn_exp2f(d);
        }
    }
    BAR_SIGNAL();

    // ---- main scan: steps 1..1022 in pairs, step 1023 peeled ----
    for (int t = 1; t <= S_ - 3; t += 2) {
        STEP_BODY(t,     sh_exp[0], sh_exp[1], sh_off[0], sh_off[1],
                  eA, mA, eB, mB, gA, gB);
        STEP_BODY(t + 1, sh_exp[1], sh_exp[0], sh_off[1], sh_off[0],
                  eB, mB, eA, mA, gB, gA);
    }
    {   // peeled final step t = S_-1 = 1023 (no publish / no prefetch)
        BAR_WAIT();
        const _Float16* abase = &sh_exp[0][n * STRIDE + sub * 8];
        FragU a0, a1, a2, a3;
        a0.h[0] = *(const v8h*)(abase);       a0.h[1] = *(const v8h*)(abase + 16);
        a1.h[0] = *(const v8h*)(abase + 32);  a1.h[1] = *(const v8h*)(abase + 48);
        a2.h[0] = *(const v8h*)(abase + 64);  a2.h[1] = *(const v8h*)(abase + 80);
        a3.h[0] = *(const v8h*)(abase + 96);  a3.h[1] = *(const v8h*)(abase + 112);
        v8f zz0 = {}, zz1 = {};
        v8f acc0 = __builtin_amdgcn_wmma_f32_16x16x32_f16(
            false, a0.v, false, bfrag[0].v, (short)0, zz0, false, false);
        v8f acc1 = __builtin_amdgcn_wmma_f32_16x16x32_f16(
            false, a1.v, false, bfrag[1].v, (short)0, zz1, false, false);
        acc0 = __builtin_amdgcn_wmma_f32_16x16x32_f16(
            false, a2.v, false, bfrag[2].v, (short)0, acc0, false, false);
        acc1 = __builtin_amdgcn_wmma_f32_16x16x32_f16(
            false, a3.v, false, bfrag[3].v, (short)0, acc1, false, false);
        SCHED_PIPELINE_PEEL();
#pragma unroll
        for (int v = 0; v < 8; ++v) {
            float lg    = __builtin_amdgcn_logf(acc0[v] + acc1[v] + EPS_);
            float fresh = fmaf(LN2, lg, gA[v] + eA[v]);
            alpha[v]    = fmaf(mA[v], fresh - alpha[v], alpha[v]);
        }
    }

    // ---- final logsumexp_j(alpha + end) per batch row (exact, once) ----
    float s[8], M[8];
#pragma unroll
    for (int v = 0; v < 8; ++v) s[v] = alpha[v] + endv;   // pads -> -inf

    // max phase
#pragma unroll
    for (int v = 0; v < 8; ++v) {
        float x = s[v];
        x = fmaxf(x, __shfl_xor(x, 1, 32));
        x = fmaxf(x, __shfl_xor(x, 2, 32));
        x = fmaxf(x, __shfl_xor(x, 4, 32));
        x = fmaxf(x, __shfl_xor(x, 8, 32));
        if (n == 0) {
            sh_red[(v + 8 * sub) * 8 + w] = x;
            if (w == 0) sh_red[(v + 8 * sub) * 8 + 7] = NEG_INF;
        }
    }
    __syncthreads();
#pragma unroll
    for (int v = 0; v < 8; ++v) {
        const float* pr = &sh_red[(v + 8 * sub) * 8];
        M[v] = fmaxf(fmaxf(fmaxf(pr[0], pr[1]), fmaxf(pr[2], pr[3])),
                     fmaxf(fmaxf(pr[4], pr[5]), fmaxf(pr[6], pr[7])));
    }
    __syncthreads();                    // sh_red reuse for sums

    // sum phase
#pragma unroll
    for (int v = 0; v < 8; ++v) {
        float x = __expf(s[v] - M[v]);  // exp(-inf)=0 handles padding
        x += __shfl_xor(x, 1, 32);
        x += __shfl_xor(x, 2, 32);
        x += __shfl_xor(x, 4, 32);
        x += __shfl_xor(x, 8, 32);
        if (n == 0) {
            sh_red[(v + 8 * sub) * 8 + w] = x;
            if (w == 0) sh_red[(v + 8 * sub) * 8 + 7] = 0.0f;
        }
    }
    __syncthreads();
    if (w == 0 && n == 0) {             // lanes 0 and 16: 8 rows each
        float tot = 0.0f;
#pragma unroll
        for (int v = 0; v < 8; ++v) {
            const float* pr = &sh_red[(v + 8 * sub) * 8];
            float sm = ((pr[0] + pr[1]) + (pr[2] + pr[3])) +
                       ((pr[4] + pr[5]) + (pr[6] + pr[7]));
            tot += M[v] + __logf(sm);
        }
        atomicAdd(out, -tot);           // minus log-denominator
    }
}

// ---------------------------------------------------------------------------
// Kernel 3: joint (numerator) score per batch row; adds +score_b into out[0].
// ---------------------------------------------------------------------------
__global__ __launch_bounds__(256) void crf_numer(
    const float* __restrict__ inputs, const float* __restrict__ trans,
    const float* __restrict__ start_t, const float* __restrict__ end_t,
    const int* __restrict__ tags, const int* __restrict__ mask,
    float* __restrict__ out) {
    __shared__ float red[256];
    const int b = blockIdx.x, tid = threadIdx.x;
    const int* tg = tags + (size_t)b * S_;
    const int* mk = mask + (size_t)b * S_;
    const float* lg = inputs + (size_t)b * S_ * T_;

    float acc = 0.0f, cnt = 0.0f;
    for (int t = tid; t < S_; t += 256) {
        float mf = (float)mk[t];
        cnt += mf;
        if (t < S_ - 1) {
            float mf1 = (float)mk[t + 1];
            int t0 = tg[t], t1 = tg[t + 1];
            acc += trans[t0 * T_ + t1] * mf1 + lg[(size_t)t * T_ + t0] * mf;
        }
    }
    red[tid] = acc;
    __syncthreads();
    for (int s2 = 128; s2 > 0; s2 >>= 1) {
        if (tid < s2) red[tid] += red[tid + s2];
        __syncthreads();
    }
    float accT = red[0];
    __syncthreads();
    red[tid] = cnt;
    __syncthreads();
    for (int s2 = 128; s2 > 0; s2 >>= 1) {
        if (tid < s2) red[tid] += red[tid + s2];
        __syncthreads();
    }
    if (tid == 0) {
        int last = (int)(red[0] + 0.5f) - 1;
        if (last < 0) last = 0;
        int lt = tg[last];
        float sc = accT + start_t[tg[0]] + end_t[lt] +
                   lg[(size_t)(S_ - 1) * T_ + lt] * (float)mk[S_ - 1];
        atomicAdd(out, sc);
    }
}

// ---------------------------------------------------------------------------
extern "C" void kernel_launch(void* const* d_in, const int* in_sizes, int n_in,
                              void* d_out, int out_size, void* d_ws, size_t ws_size,
                              hipStream_t stream) {
    const float* inputs = (const float*)d_in[0];
    const float* trans  = (const float*)d_in[1];
    const float* startt = (const float*)d_in[2];
    const float* endt   = (const float*)d_in[3];
    const int*   tags   = (const int*)d_in[4];
    const int*   maskp  = (const int*)d_in[5];
    float* out = (float*)d_out;
    _Float16* et = (_Float16*)d_ws;     // 128*128*2 = 32 KB scratch

    hipMemsetAsync(out, 0, sizeof(float), stream);
    crf_prep_et<<<(TP * TP + 255) / 256, 256, 0, stream>>>(trans, et);
    crf_numer<<<B_, 256, 0, stream>>>(inputs, trans, startt, endt, tags, maskp, out);
    crf_forward<<<B_ / 16, NW * 32, 0, stream>>>(inputs, startt, endt, maskp, et, out);
}
